// CrossTransAttention_45681272160409
// MI455X (gfx1250) — matrix operand order repaired and checked by reading, not compiled, not run on hardware
//
#include <hip/hip_runtime.h>
#include <hip/hip_bf16.h>

typedef __attribute__((ext_vector_type(16))) __bf16       v16bf;
typedef __attribute__((ext_vector_type(8)))  __bf16       v8bf;
typedef __attribute__((ext_vector_type(8)))  float        v8f;
typedef __attribute__((ext_vector_type(4)))  unsigned int v4u;
typedef __attribute__((ext_vector_type(8)))  int          v8i;
typedef __attribute__((ext_vector_type(4)))  int          v4i;

#define NPX 65536   // H*W
#define CDIM 192
#define HEADS 6
#define KP 200      // padded K stride (bf16 elems) for LDS B tile
#define DWROWS 16   // output rows per dwconv block

// ---------------------------------------------------------------------------
// zero small accumulators (sqsum_q, sqsum_k, G) — contiguous region
// ---------------------------------------------------------------------------
__global__ __launch_bounds__(256) void zero_kernel(float* __restrict__ p, int n) {
    int i = blockIdx.x * 256 + threadIdx.x;
    if (i < n) p[i] = 0.f;
}

// ---------------------------------------------------------------------------
// TDM: load `nrows` contiguous bf16 rows of 256 elems (row stride 256 elems)
// from global into LDS at byte offset lds_off. 2D D# per ISA 08_async_tensor §8.
// Must be executed by one wave only (EXEC ignored by TDM, but issue is per-wave).
// ---------------------------------------------------------------------------
__device__ static inline void tdm_load_rows(const __bf16* gsrc, unsigned int lds_off,
                                            int nrows) {
#if __has_builtin(__builtin_amdgcn_tensor_load_to_lds)
    unsigned long long ga = (unsigned long long)(uintptr_t)gsrc;
    v4u g0;
    g0[0] = 1u;                                   // count=1, is_restore=0, no gather
    g0[1] = lds_off;                              // lds_addr
    g0[2] = (unsigned int)(ga & 0xFFFFFFFFu);     // global_addr[31:0]
    g0[3] = (unsigned int)((ga >> 32) & 0x1FFFFFFu) | (2u << 30);  // addr[56:32], type=2
    v8i g1;
    g1[0] = (int)(1u << 16);                      // workgroup_mask=0, data_size=1 (2B)
    g1[1] = (int)(256u << 16);                    // tensor_dim0[15:0]=256 in bits[63:48]
    g1[2] = (int)(((unsigned int)nrows) << 16);   // tensor_dim0 hi=0, tensor_dim1=nrows
    g1[3] = (int)(256u << 16);                    // tensor_dim1 hi=0, tile_dim0=256
    g1[4] = (int)((unsigned int)nrows);           // tile_dim1=nrows, tile_dim2=0
    g1[5] = 256;                                  // tensor_dim0_stride[31:0]=256
    g1[6] = 0;                                    // stride hi, tensor_dim1_stride lo
    g1[7] = 0;
    v4i gz = {0, 0, 0, 0};
#if defined(__clang_major__) && (__clang_major__ >= 23)
    v8i gz8 = {0, 0, 0, 0, 0, 0, 0, 0};
    __builtin_amdgcn_tensor_load_to_lds(g0, g1, gz, gz, gz8, 0);
#else
    __builtin_amdgcn_tensor_load_to_lds(g0, g1, gz, gz, 0);
#endif
    __builtin_amdgcn_s_wait_tensorcnt(0);
#else
    // fallback: stub, performs no copy
    __bf16* lds = (__bf16*)(uintptr_t)0;  // unreachable on gfx1250 toolchains
    (void)gsrc; (void)lds_off; (void)nrows; (void)lds;
#endif
}

// ---------------------------------------------------------------------------
// GEMM: Y[b][m][n] = sum_k A[m][k] * X[b][k][n] + bias[m]   (K = 192)
//   A_BF16: A bf16 (per-batch stride aStride) else fp32 shared
//   X_BF16: X bf16 else fp32 (converted while staging to LDS)
//   OUT_F32: fp32 out else bf16
// block = 256 threads (8 waves); block covers all MT rows x 64 columns.
// ---------------------------------------------------------------------------
template <int MT, bool A_BF16, bool X_BF16, bool OUT_F32>
__global__ __launch_bounds__(256) void gemm_conv1x1(
    const void* __restrict__ Wp, const void* __restrict__ Xp,
    const float* __restrict__ bias, void* __restrict__ Yp, int aStride)
{
    __shared__ __bf16 Blds[64 * KP];

    const int b  = blockIdx.y;
    const int n0 = blockIdx.x * 64;
    const int t  = threadIdx.x;

    if (X_BF16) {
        const __bf16* X = (const __bf16*)Xp + (size_t)b * CDIM * NPX;
        for (int i = t; i < CDIM * 64; i += 256) {
            int k = i >> 6, n = i & 63;
            Blds[n * KP + k] = X[(size_t)k * NPX + n0 + n];
        }
    } else {
        const float* X = (const float*)Xp + (size_t)b * CDIM * NPX;
        for (int i = t; i < CDIM * 64; i += 256) {
            int k = i >> 6, n = i & 63;
            Blds[n * KP + k] = (__bf16)X[(size_t)k * NPX + n0 + n];
        }
    }
    __syncthreads();

    const int wave = t >> 5;
    const int lane = t & 31;
    const int lrow = lane & 15;
    const int lhi  = lane >> 4;

    for (int idx = wave; idx < (MT / 16) * 4; idx += 8) {
        const int mt = idx >> 2;
        const int nt = idx & 3;
        const int m0 = mt * 16;
        v8f acc = {};

#pragma unroll
        for (int kb = 0; kb < CDIM; kb += 32) {
            const int m  = m0 + lrow;
            const int k0 = kb + lhi * 8;
            v16bf a;
            if (A_BF16) {
                const __bf16* A = (const __bf16*)Wp + (size_t)b * aStride;
                v8bf a0 = *(const v8bf*)(A + (size_t)m * CDIM + k0);
                v8bf a1 = *(const v8bf*)(A + (size_t)m * CDIM + k0 + 16);
#pragma unroll
                for (int j = 0; j < 8; ++j) { a[j] = a0[j]; a[8 + j] = a1[j]; }
            } else {
                const float* A = (const float*)Wp;
                const float4* p0 = (const float4*)(A + (size_t)m * CDIM + k0);
                const float4* p1 = (const float4*)(A + (size_t)m * CDIM + k0 + 16);
                float4 w0 = p0[0], w1 = p0[1], w2 = p1[0], w3 = p1[1];
                a[0]=(__bf16)w0.x; a[1]=(__bf16)w0.y; a[2]=(__bf16)w0.z; a[3]=(__bf16)w0.w;
                a[4]=(__bf16)w1.x; a[5]=(__bf16)w1.y; a[6]=(__bf16)w1.z; a[7]=(__bf16)w1.w;
                a[8]=(__bf16)w2.x; a[9]=(__bf16)w2.y; a[10]=(__bf16)w2.z; a[11]=(__bf16)w2.w;
                a[12]=(__bf16)w3.x; a[13]=(__bf16)w3.y; a[14]=(__bf16)w3.z; a[15]=(__bf16)w3.w;
            }
            const __bf16* bp = &Blds[(nt * 16 + lrow) * KP + kb + lhi * 8];
            v8bf b0 = *(const v8bf*)bp;
            v8bf b1 = *(const v8bf*)(bp + 16);
            v16bf bm;
#pragma unroll
            for (int j = 0; j < 8; ++j) { bm[j] = b0[j]; bm[8 + j] = b1[j]; }

            acc = __builtin_amdgcn_wmma_f32_16x16x32_bf16(
                false, a, false, bm, (short)0, acc, false, false);
        }

#pragma unroll
        for (int r = 0; r < 8; ++r) {
            const int m = m0 + r + lhi * 8;
            const size_t n = (size_t)n0 + nt * 16 + lrow;
            const float v = acc[r] + bias[m];
            const size_t o = (size_t)b * MT * NPX + (size_t)m * NPX + n;
            if (OUT_F32) ((float*)Yp)[o] = v;
            else         ((__bf16*)Yp)[o] = (__bf16)v;
        }
    }
}

// ---------------------------------------------------------------------------
// depthwise 3x3 (pad=1): TDM stages an 18x256 bf16 halo tile into LDS,
// each thread computes DWROWS outputs in its column; fp32 accumulate,
// bf16 out, optional per-channel sum-of-squares (LDS reduce + 1 atomic).
// grid = (256/DWROWS, Ct, B), block = 256
// ---------------------------------------------------------------------------
__global__ __launch_bounds__(256) void dwconv_kernel(
    const __bf16* __restrict__ src, const float* __restrict__ wt,
    const float* __restrict__ bias, __bf16* __restrict__ outA,
    __bf16* __restrict__ outB, float* __restrict__ sqsum,
    int Ct, int split, int sqLimit)
{
    __shared__ __bf16 tile[(DWROWS + 2) * 256];
    __shared__ float red[256];

    const int rb = blockIdx.x;
    const int c  = blockIdx.y;
    const int b  = blockIdx.z;
    const int t  = threadIdx.x;
    const int w  = t;
    const int h0 = rb * DWROWS;
    const size_t plane = ((size_t)b * Ct + c) * NPX;

    // zero-fill halo rows that fall outside the image (TDM won't touch them)
    if (h0 == 0)            tile[t] = (__bf16)0.f;
    if (h0 + DWROWS == 256) tile[(DWROWS + 1) * 256 + t] = (__bf16)0.f;

    int lo = h0 - 1;         if (lo < 0) lo = 0;
    int hi = h0 + DWROWS;    if (hi > 255) hi = 255;
    const int nrows = hi - lo + 1;

    if (t < 32) {
        unsigned int lds_off =
            (unsigned int)(uintptr_t)(void*)&tile[(lo - (h0 - 1)) * 256];
        tdm_load_rows(src + plane + (size_t)lo * 256, lds_off, nrows);
    }
    __syncthreads();

    float k[9];
#pragma unroll
    for (int i = 0; i < 9; ++i) k[i] = wt[c * 9 + i];
    const float bs = bias[c];

    float sq = 0.f;
#pragma unroll
    for (int r = 0; r < DWROWS; ++r) {
        const __bf16* r0 = &tile[r * 256];
        const __bf16* r1 = r0 + 256;
        const __bf16* r2 = r1 + 256;
        float acc = bs;
        if (w > 0)
            acc += k[0] * (float)r0[w-1] + k[3] * (float)r1[w-1] + k[6] * (float)r2[w-1];
        acc += k[1] * (float)r0[w] + k[4] * (float)r1[w] + k[7] * (float)r2[w];
        if (w < 255)
            acc += k[2] * (float)r0[w+1] + k[5] * (float)r1[w+1] + k[8] * (float)r2[w+1];

        const size_t n = (size_t)(h0 + r) * 256 + w;
        if (c < split) outA[((size_t)b * split + c) * NPX + n] = (__bf16)acc;
        else           outB[((size_t)b * (Ct - split) + (c - split)) * NPX + n] = (__bf16)acc;
        sq += acc * acc;
    }

    if (sqsum != nullptr && c < sqLimit) {
        red[t] = sq;
        __syncthreads();
#pragma unroll
        for (int s = 128; s > 0; s >>= 1) {
            if (t < s) red[t] += red[t + s];
            __syncthreads();
        }
        if (t == 0) atomicAdd(&sqsum[b * CDIM + c], red[0]);
    }
}

// ---------------------------------------------------------------------------
// Gram: G[b][h][c][d] += sum_k q[h*32+c][k] * k[h*32+d][k]  (K split in slabs)
// ---------------------------------------------------------------------------
__global__ __launch_bounds__(256) void gram_kernel(
    const __bf16* __restrict__ Q, const __bf16* __restrict__ Kmat,
    float* __restrict__ G)
{
    __shared__ float Gp[1024];
    const int slab = blockIdx.x, h = blockIdx.y, b = blockIdx.z;
    const int t = threadIdx.x;
    const int wave = t >> 5, lane = t & 31, lrow = lane & 15, lhi = lane >> 4;

    for (int i = t; i < 1024; i += 256) Gp[i] = 0.f;
    __syncthreads();

    const size_t base = ((size_t)b * CDIM + h * 32) * NPX;
    const int kstart = slab * 4096 + wave * 512;

    v8f acc[2][2];
#pragma unroll
    for (int i = 0; i < 2; ++i)
#pragma unroll
        for (int j = 0; j < 2; ++j) acc[i][j] = (v8f){};

    for (int kb = kstart; kb < kstart + 512; kb += 32) {
        v16bf aq[2], bk[2];
#pragma unroll
        for (int mt = 0; mt < 2; ++mt) {
            const __bf16* p = Q + base + (size_t)(mt * 16 + lrow) * NPX + kb + lhi * 8;
            v8bf x0 = *(const v8bf*)p;
            v8bf x1 = *(const v8bf*)(p + 16);
#pragma unroll
            for (int j = 0; j < 8; ++j) { aq[mt][j] = x0[j]; aq[mt][8 + j] = x1[j]; }
        }
#pragma unroll
        for (int nt = 0; nt < 2; ++nt) {
            const __bf16* p = Kmat + base + (size_t)(nt * 16 + lrow) * NPX + kb + lhi * 8;
            v8bf y0 = *(const v8bf*)p;
            v8bf y1 = *(const v8bf*)(p + 16);
#pragma unroll
            for (int j = 0; j < 8; ++j) { bk[nt][j] = y0[j]; bk[nt][8 + j] = y1[j]; }
        }
#pragma unroll
        for (int mt = 0; mt < 2; ++mt)
#pragma unroll
            for (int nt = 0; nt < 2; ++nt)
                acc[mt][nt] = __builtin_amdgcn_wmma_f32_16x16x32_bf16(
                    false, aq[mt], false, bk[nt], (short)0, acc[mt][nt], false, false);
    }

#pragma unroll
    for (int mt = 0; mt < 2; ++mt)
#pragma unroll
        for (int nt = 0; nt < 2; ++nt)
#pragma unroll
            for (int r = 0; r < 8; ++r)
                atomicAdd(&Gp[(mt * 16 + r + lhi * 8) * 32 + nt * 16 + lrow],
                          acc[mt][nt][r]);
    __syncthreads();

    float* g = G + ((size_t)b * HEADS + h) * 1024;
    for (int i = t; i < 1024; i += 256) atomicAdd(&g[i], Gp[i]);
}

// ---------------------------------------------------------------------------
// softmax of scaled Gram + fold with proj_w -> Mfold[b][o][h*32+e] (bf16)
// ---------------------------------------------------------------------------
__global__ __launch_bounds__(256) void softmax_fold_kernel(
    const float* __restrict__ G, const float* __restrict__ sq_q,
    const float* __restrict__ sq_k, const float* __restrict__ temp,
    const float* __restrict__ projW, __bf16* __restrict__ Mfold)
{
    __shared__ float attn[HEADS * 1024];
    __shared__ float rq[CDIM], rk[CDIM];
    const int b = blockIdx.x;
    const int t = threadIdx.x;

    for (int i = t; i < CDIM; i += 256) {
        rq[i] = 1.f / fmaxf(sqrtf(sq_q[b * CDIM + i]), 1e-12f);
        rk[i] = 1.f / fmaxf(sqrtf(sq_k[b * CDIM + i]), 1e-12f);
    }
    __syncthreads();

    if (t < CDIM) {
        const int hh = t / 32, cr = t % 32;
        const float* g = G + ((size_t)b * HEADS + hh) * 1024 + cr * 32;
        const float tv = temp[hh];
        float l[32], mx = -3.4e38f;
#pragma unroll
        for (int d = 0; d < 32; ++d) {
            l[d] = g[d] * rq[hh * 32 + cr] * rk[hh * 32 + d] * tv;
            mx = fmaxf(mx, l[d]);
        }
        float s = 0.f;
#pragma unroll
        for (int d = 0; d < 32; ++d) { l[d] = expf(l[d] - mx); s += l[d]; }
        const float inv = 1.f / s;
#pragma unroll
        for (int d = 0; d < 32; ++d) attn[hh * 1024 + cr * 32 + d] = l[d] * inv;
    }
    __syncthreads();

    for (int i = t; i < CDIM * CDIM; i += 256) {
        const int o = i / CDIM, ce = i % CDIM;
        const int hh = ce / 32, e = ce % 32;
        const float* prow = projW + o * CDIM + hh * 32;
        float acc = 0.f;
#pragma unroll
        for (int d = 0; d < 32; ++d) acc += prow[d] * attn[hh * 1024 + d * 32 + e];
        Mfold[((size_t)b * CDIM + o) * CDIM + ce] = (__bf16)acc;
    }
}

// ---------------------------------------------------------------------------
extern "C" void kernel_launch(void* const* d_in, const int* in_sizes, int n_in,
                              void* d_out, int out_size, void* d_ws, size_t ws_size,
                              hipStream_t stream) {
    const float* feat_guide = (const float*)d_in[0];
    const float* feat_op    = (const float*)d_in[1];
    const float* q_w    = (const float*)d_in[2];
    const float* q_b    = (const float*)d_in[3];
    const float* kv_w   = (const float*)d_in[4];
    const float* kv_b   = (const float*)d_in[5];
    const float* q_dw_w = (const float*)d_in[6];
    const float* q_dw_b = (const float*)d_in[7];
    const float* kv_dw_w= (const float*)d_in[8];
    const float* kv_dw_b= (const float*)d_in[9];
    const float* proj_w = (const float*)d_in[10];
    const float* proj_b = (const float*)d_in[11];
    const float* temperature = (const float*)d_in[12];

    char* ws = (char*)d_ws;
    __bf16* qpre  = (__bf16*)(ws);                      // [2][192][N] bf16  48MB
    __bf16* kvpre = (__bf16*)(ws + 50331648ull);        // [2][384][N] bf16  96MB
    __bf16* qb    = (__bf16*)(ws + 150994944ull);       // [2][192][N] bf16  48MB
    __bf16* kb    = (__bf16*)(ws + 201326592ull);       // [2][192][N] bf16  48MB
    __bf16* vb    = (__bf16*)(ws + 251658240ull);       // [2][192][N] bf16  48MB
    float*  sqq   = (float*)(ws + 301989888ull);        // [2][192]
    float*  sqk   = sqq + 384;                          // [2][192]
    float*  G     = sqk + 384;                          // [2][6][32][32]
    __bf16* Mf    = (__bf16*)(G + 2 * HEADS * 1024);    // [2][192][192] bf16

    zero_kernel<<<dim3(51), dim3(256), 0, stream>>>(sqq, 384 + 384 + 2 * HEADS * 1024);

    gemm_conv1x1<192, false, false, false><<<dim3(NPX / 64, 2), dim3(256), 0, stream>>>(
        q_w, feat_guide, q_b, qpre, 0);

    gemm_conv1x1<384, false, false, false><<<dim3(NPX / 64, 2), dim3(256), 0, stream>>>(
        kv_w, feat_op, kv_b, kvpre, 0);

    dwconv_kernel<<<dim3(256 / DWROWS, 192, 2), dim3(256), 0, stream>>>(
        qpre, q_dw_w, q_dw_b, qb, nullptr, sqq, 192, 192, 192);

    dwconv_kernel<<<dim3(256 / DWROWS, 384, 2), dim3(256), 0, stream>>>(
        kvpre, kv_dw_w, kv_dw_b, kb, vb, sqk, 384, 192, 192);

    gram_kernel<<<dim3(16, HEADS, 2), dim3(256), 0, stream>>>(qb, kb, G);

    softmax_fold_kernel<<<dim3(2), dim3(256), 0, stream>>>(
        G, sqq, sqk, temperature, proj_w, Mf);

    gemm_conv1x1<192, true, true, true><<<dim3(NPX / 64, 2), dim3(256), 0, stream>>>(
        Mf, vb, proj_b, d_out, CDIM * CDIM);
}
